// MinimalReservoir_23837068493313
// MI455X (gfx1250) — compile-verified, matrix operand order checked by reading, not run on hardware
//
#include <hip/hip_runtime.h>
#include <hip/hip_bf16.h>

#define SEQ 4096
#define INP 512
#define RES 2048

#define NWG 32
#define RPW 64            // W_res rows per workgroup
#define PITCH 2052        // f16 per LDS row: dword stride 1026 ≡ 2 (mod 64) -> conflict-free

typedef __attribute__((ext_vector_type(16))) _Float16 v16h;
typedef __attribute__((ext_vector_type(8)))  float    v8f;
typedef __attribute__((ext_vector_type(4)))  unsigned int uint4v;
typedef __attribute__((ext_vector_type(8)))  int          int8v;
typedef __attribute__((ext_vector_type(4)))  int          int4v;

// gcc-style vector for the async-to-LDS builtin's pointer params
// (diagnostic showed: 'int __attribute__((vector_size(16))) __device__ *')
typedef int gint4 __attribute__((vector_size(16)));
#define GPTR_I4(p) ((__attribute__((address_space(1))) gint4*)(p))
#define LPTR_I4(p) ((__attribute__((address_space(3))) gint4*)(p))
#define AS_LOCAL(p) ((__attribute__((address_space(3))) void*)(p))

// ---------------------------------------------------------------------------
// Phase 1: pre = X @ W_in^T into out rows 1..SEQ (fp32), via f16 WMMA.
// One wave per 16x16 tile, K-loop of v_wmma_f32_16x16x32_f16 (ISA 7.12.2
// fragment layouts; C layout: VGPR r -> M = r + 8h, N = lane&15).
// ---------------------------------------------------------------------------
__global__ __launch_bounds__(256) void proj_gemm(const float* __restrict__ X,
                                                 const float* __restrict__ Win,
                                                 float* __restrict__ out) {
  const int lane = threadIdx.x & 31;
  const int wv   = threadIdx.x >> 5;
  const int gw   = blockIdx.x * 8 + wv;       // 0..32767 tiles
  const int tn   = gw & 127;                  // 2048/16 n-tiles
  const int tm   = gw >> 7;                   // 4096/16 m-tiles
  const int h    = lane >> 4;
  const int l16  = lane & 15;

  const float* arow = X   + (size_t)(tm * 16 + l16) * INP;
  const float* brow = Win + (size_t)(tn * 16 + l16) * INP;

  v8f c = {};
  for (int k0 = 0; k0 < INP; k0 += 32) {
    __builtin_prefetch(arow + k0 + 64, 0, 1);
    __builtin_prefetch(brow + k0 + 64, 0, 1);
    v16h a, b;
#pragma unroll
    for (int i = 0; i < 8; ++i) {
      a[i]     = (_Float16)arow[k0 + 8 * h + i];
      a[i + 8] = (_Float16)arow[k0 + 16 + 8 * h + i];
      b[i]     = (_Float16)brow[k0 + 8 * h + i];
      b[i + 8] = (_Float16)brow[k0 + 16 + 8 * h + i];
    }
    c = __builtin_amdgcn_wmma_f32_16x16x32_f16(false, a, false, b,
                                               (short)0, c, false, false);
  }
#pragma unroll
  for (int r = 0; r < 8; ++r) {
    const int m = tm * 16 + r + 8 * h;
    const int n = tn * 16 + l16;
    out[(size_t)(m + 1) * RES + n] = c[r];   // rows 1..SEQ hold pre
  }
}

// ---------------------------------------------------------------------------
// Phase 2: persistent scan, W_res resident in LDS as f16 (257 KB/WGP).
// Per step: f16xF32 fma-mix dots (state reads wave-uniform -> LDS broadcast;
// W row dword-stride 1026 = 2 mod 64 -> 32 rows hit 32 distinct banks),
// LDS reduction, tanh, in-place update of out row t, monotonic agent-scope
// barrier, then re-broadcast state into LDS via TDM / async-to-LDS.
// ---------------------------------------------------------------------------
__global__ __launch_bounds__(256) void reservoir_scan(const float* __restrict__ Wres,
                                                      const float* __restrict__ s0,
                                                      float* __restrict__ out,
                                                      unsigned* __restrict__ bar) {
  __shared__ _Float16 wl[RPW * PITCH];   // ~257 KB of the 320 KB WGP LDS
  __shared__ float    sl[RES];           // current state, 8 KB
  __shared__ float    red[256];

  const int tid   = threadIdx.x;
  const int g     = blockIdx.x;
  const int rbase = g * RPW;

  // Stage this WG's W_res slice, fp32 -> f16, into padded LDS (float4 loads).
  for (int i = tid * 4; i < RPW * RES; i += 256 * 4) {
    const int r = i >> 11;          // / RES
    const int k = i & (RES - 1);
    const float4 w4 = *(const float4*)(Wres + (size_t)(rbase + r) * RES + k);
    _Float16* dst = wl + r * PITCH + k;
    dst[0] = (_Float16)w4.x; dst[1] = (_Float16)w4.y;
    dst[2] = (_Float16)w4.z; dst[3] = (_Float16)w4.w;
  }

  // Initial state into LDS (async global->LDS if available) + row 0 output.
#if __has_builtin(__builtin_amdgcn_global_load_async_to_lds_b128)
  {
    float* src = const_cast<float*>(s0) + tid * 8;
    __builtin_amdgcn_global_load_async_to_lds_b128(GPTR_I4(src),
                                                   LPTR_I4(sl + tid * 8), 0, 0);
    __builtin_amdgcn_global_load_async_to_lds_b128(GPTR_I4(src + 4),
                                                   LPTR_I4(sl + tid * 8 + 4), 0, 0);
    asm volatile("s_wait_asynccnt 0x0" ::: "memory");
  }
  if (g == 0)
    for (int i = tid; i < RES; i += 256) out[i] = s0[i];
#else
  for (int i = tid; i < RES; i += 256) {
    const float v = s0[i];
    sl[i] = v;
    if (g == 0) out[i] = v;
  }
#endif
  __syncthreads();

  const float inv = 0.022097086912079608f;   // 1/sqrt(2048)
  const int row = tid & (RPW - 1);           // wave-contiguous rows
  const int c   = tid >> 6;                  // k-chunk, uniform per wave

  for (int t = 1; t <= SEQ; ++t) {
    float acc = 0.f;
    const _Float16* wr = wl + row * PITCH + c * 8;
    const float*    sp = sl + c * 8;
#pragma unroll 4
    for (int j = 0; j < 64; ++j) {
#pragma unroll
      for (int i = 0; i < 8; ++i)
        acc = fmaf((float)wr[32 * j + i], sp[32 * j + i], acc);
    }
    red[tid] = acc;
    __syncthreads();
    if (tid < RPW) {
      const float s = red[tid] + red[tid + 64] + red[tid + 128] + red[tid + 192];
      const size_t idx = (size_t)t * RES + (rbase + tid);
      const float pre = out[idx];              // GEMM result, read-once
      out[idx] = tanhf(pre + s) * inv;         // in-place: pre -> state
    }
    __syncthreads();
    if (tid == 0) {
      __threadfence();
      __hip_atomic_fetch_add(bar, 1u, __ATOMIC_RELEASE, __HIP_MEMORY_SCOPE_AGENT);
      const unsigned target = (unsigned)t * NWG;
      while (__hip_atomic_load(bar, __ATOMIC_ACQUIRE, __HIP_MEMORY_SCOPE_AGENT) < target)
        __builtin_amdgcn_s_sleep(1);
    }
    __syncthreads();

    if (t < SEQ) {
      // Re-broadcast the freshly written state row t into LDS.
#if __has_builtin(__builtin_amdgcn_tensor_load_to_lds)
      if (tid < 32) {   // wave 0 only: TDM issues per-wave, EXEC ignored
        const unsigned long long ga =
            (unsigned long long)(uintptr_t)(out + (size_t)t * RES);
        const unsigned lds_sl = (unsigned)(uintptr_t)AS_LOCAL(sl);
        uint4v g0;
        g0.x = 1u;                                        // count=1 (valid D#)
        g0.y = lds_sl;                                    // lds_addr
        g0.z = (unsigned)(ga & 0xFFFFFFFFu);              // global_addr[31:0]
        g0.w = (unsigned)((ga >> 32) & 0x01FFFFFFu)       // global_addr[56:32]
             | (2u << 30);                                // type=2 (image)
        int8v g1;
        g1[0] = (int)(2u << 16);                          // data_size=4B, mask=0
        g1[1] = (int)((RES & 0xFFFFu) << 16);             // tensor_dim0 lo16
        g1[2] = (int)((RES >> 16) | (1u << 16));          // dim0 hi16 | dim1 lo16
        g1[3] = (int)((unsigned)RES << 16);               // dim1 hi16=0 | tile_dim0
        g1[4] = 1;                                        // tile_dim1=1, tile_dim2=0
        g1[5] = RES;                                      // dim0_stride lo32
        g1[6] = (int)((unsigned)RES << 16);               // stride hi16=0 | dim1_stride lo16
        g1[7] = 0;
        const int4v z4 = {};
#if __has_include(<hip/amd_detail/amd_gfx1250_TDM.h>)
        const int8v z8 = {};
        __builtin_amdgcn_tensor_load_to_lds(g0, g1, z4, z4, z8, 0);  // clang-23 form
#else
        __builtin_amdgcn_tensor_load_to_lds(g0, g1, z4, z4, 0);      // ROCm 7.2 form
#endif
        __builtin_amdgcn_s_wait_tensorcnt((short)0);
      }
#elif __has_builtin(__builtin_amdgcn_global_load_async_to_lds_b128)
      {
        float* src = out + (size_t)t * RES + tid * 8;
        __builtin_amdgcn_global_load_async_to_lds_b128(GPTR_I4(src),
                                                       LPTR_I4(sl + tid * 8), 0, 0);
        __builtin_amdgcn_global_load_async_to_lds_b128(GPTR_I4(src + 4),
                                                       LPTR_I4(sl + tid * 8 + 4), 0, 0);
        asm volatile("s_wait_asynccnt 0x0" ::: "memory");
      }
#else
      for (int i = tid; i < RES; i += 256)
        sl[i] = out[(size_t)t * RES + i];
#endif
      __syncthreads();
    }
  }
}

extern "C" void kernel_launch(void* const* d_in, const int* in_sizes, int n_in,
                              void* d_out, int out_size, void* d_ws, size_t ws_size,
                              hipStream_t stream) {
  (void)in_sizes; (void)n_in; (void)out_size; (void)ws_size;
  const float* X    = (const float*)d_in[0];   // (4096, 512)
  const float* s0   = (const float*)d_in[1];   // (2048,)
  const float* Win  = (const float*)d_in[2];   // (2048, 512)
  const float* Wres = (const float*)d_in[3];   // (2048, 2048)
  float* out = (float*)d_out;                  // (4097, 2048)
  unsigned* bar = (unsigned*)d_ws;

  (void)hipMemsetAsync(bar, 0, 256, stream);                // barrier counter = 0
  proj_gemm<<<4096, 256, 0, stream>>>(X, Win, out);         // rows 1..4096 = pre
  reservoir_scan<<<NWG, 256, 0, stream>>>(Wres, s0, out, bar);
}
